// RelativeMultiheadSelfAttention_13108240188131
// MI455X (gfx1250) — matrix operand order repaired and checked by reading, not compiled, not run on hardware
//
#include <hip/hip_runtime.h>

// ---------------------------------------------------------------------------
// Transformer-XL relative multi-head self-attention for MI455X (gfx1250).
// - bf16 WMMA (v_wmma_f32_16x16x32_bf16) for every matmul, fp32 accumulation
// - flash-attention streaming over causal key tiles
// - rel_shift realized as banded pos GEMM + LDS shear
// - async global->LDS copies (ASYNCcnt) with double-buffered tiles
// ---------------------------------------------------------------------------

#define DEV __device__ __forceinline__

typedef __attribute__((ext_vector_type(16))) __bf16 v16bf;
typedef __attribute__((ext_vector_type(8)))  float  v8f;

union Frag16 { v16bf v; unsigned u[8]; };
union Frag32 { v8f  v; float    f[8]; };

static constexpr int TT = 2048, BB = 2, DD = 768, NH = 12, HD = 64;
static constexpr float SCALE = 0.036084391824351615f; // 1/sqrt(768)
static constexpr float NEG_BIG = -3.0e38f;

DEV unsigned short f2bf(float x) {
  unsigned u = __builtin_bit_cast(unsigned, x);
  u += 0x7FFFu + ((u >> 16) & 1u);            // round-to-nearest-even
  return (unsigned short)(u >> 16);
}

// K-pair base (pairs of bf16 packed per VGPR) for A (16x32) and B (32x16)
// fragments per the CDNA5 ISA layouts (lane half h selects K half-block).
DEV int kApair(int v, int h) { return ((v & 3) << 1) + ((v >> 2) << 4) + (h << 3); }
DEV int kBpair(int v, int h) { return ((v & 3) << 1) + ((v >> 2) << 4) + (h << 3); }

DEV v8f wmma_bf16(v16bf a, v16bf b, v8f c) {
  return __builtin_amdgcn_wmma_f32_16x16x32_bf16(false, a, false, b, (short)0, c,
                                                 false, false);
}

// Async 16B global -> LDS copy, one chunk per lane (GLOBAL_LOAD_ASYNC_TO_LDS,
// tracked with ASYNCcnt; see CDNA5 ISA).
DEV void async_g2l_b128(unsigned lds_off, const void* gptr) {
  unsigned long long ga = (unsigned long long)gptr;
  asm volatile("global_load_async_to_lds_b128 %0, %1, off"
               :: "v"(lds_off), "v"(ga) : "memory");
}
DEV void wait_async0() { asm volatile("s_wait_asynccnt 0x0" ::: "memory"); }
DEV unsigned lds_addr(const void* p) { return (unsigned)(size_t)p; }

DEV float rmax16(float x) {
  x = fmaxf(x, __shfl_xor(x, 1));
  x = fmaxf(x, __shfl_xor(x, 2));
  x = fmaxf(x, __shfl_xor(x, 4));
  x = fmaxf(x, __shfl_xor(x, 8));
  return x;
}
DEV float rsum16(float x) {
  x += __shfl_xor(x, 1);
  x += __shfl_xor(x, 2);
  x += __shfl_xor(x, 4);
  x += __shfl_xor(x, 8);
  return x;
}

// ---------------------------------------------------------------------------
// fp32 -> bf16 elementwise pack
// ---------------------------------------------------------------------------
__global__ void cvt_f32_bf16(const float* __restrict__ src,
                             unsigned short* __restrict__ dst, int n) {
  int i = blockIdx.x * blockDim.x + threadIdx.x;
  if (i < n) dst[i] = f2bf(src[i]);
}

// ---------------------------------------------------------------------------
// Generic bf16 GEMM: C[M,N] (fp32) = A[M,K] * Bm[K,N], all row-major.
// Block tile 64x128, 8 waves (4x2), wave tile 16x64, K-step 64.
// A tile staged via async DMA (overlapped with the VGPR-mediated B transpose).
// ---------------------------------------------------------------------------
__global__ __launch_bounds__(256) void gemm_bf16(
    const unsigned short* __restrict__ A, int K,
    const unsigned short* __restrict__ Bm, int N,
    float* __restrict__ C, int M) {
  __shared__ alignas(16) unsigned short Al[64][72];
  __shared__ alignas(16) unsigned short BlT[128][72];

  const int tid = threadIdx.x;
  const int lane = tid & 31, w = tid >> 5;
  const int wr = w >> 1, wc = w & 1;
  const int ln = lane & 15, h = lane >> 4;
  const int M0 = blockIdx.y * 64, N0 = blockIdx.x * 128;

  Frag32 acc[4];
#pragma unroll
  for (int f = 0; f < 4; ++f)
#pragma unroll
    for (int r = 0; r < 8; ++r) acc[f].f[r] = 0.f;

  for (int k0 = 0; k0 < K; k0 += 64) {
    __syncthreads();
    // Stage A tile [64m x 64k] with async DMA: 512 x 16B chunks.
#pragma unroll
    for (int it = 0; it < 2; ++it) {
      int idx = tid + 256 * it;
      int r = idx >> 3, c8 = idx & 7;
      async_g2l_b128(lds_addr(&Al[r][c8 * 8]),
                     A + (size_t)(M0 + r) * K + k0 + c8 * 8);
    }
    // Stage B tile transposed (BlT[n][k]); load phase then store phase so the
    // backend batches the global loads under a single wait.
    unsigned bv0[8], bv1[8];
#pragma unroll
    for (int it = 0; it < 8; ++it) {
      int idx = tid + 256 * it;
      int kk = idx >> 5, c4 = idx & 31;
      const unsigned* p =
          (const unsigned*)(Bm + (size_t)(k0 + kk) * N + N0 + c4 * 4);
      bv0[it] = p[0];
      bv1[it] = p[1];
    }
#pragma unroll
    for (int it = 0; it < 8; ++it) {
      int idx = tid + 256 * it;
      int kk = idx >> 5, c4 = idx & 31;
      BlT[c4 * 4 + 0][kk] = (unsigned short)(bv0[it] & 0xFFFFu);
      BlT[c4 * 4 + 1][kk] = (unsigned short)(bv0[it] >> 16);
      BlT[c4 * 4 + 2][kk] = (unsigned short)(bv1[it] & 0xFFFFu);
      BlT[c4 * 4 + 3][kk] = (unsigned short)(bv1[it] >> 16);
    }
    wait_async0();
    __syncthreads();
#pragma unroll
    for (int ks = 0; ks < 2; ++ks) {
      Frag16 af;
#pragma unroll
      for (int v = 0; v < 8; ++v)
        af.u[v] = *(const unsigned*)&Al[16 * wr + ln][kApair(v, h) + 32 * ks];
#pragma unroll
      for (int f = 0; f < 4; ++f) {
        Frag16 bf_;
#pragma unroll
        for (int v = 0; v < 8; ++v)
          bf_.u[v] =
              *(const unsigned*)&BlT[64 * wc + 16 * f + ln][kBpair(v, h) + 32 * ks];
        acc[f].v = wmma_bf16(af.v, bf_.v, acc[f].v);
      }
    }
  }
#pragma unroll
  for (int f = 0; f < 4; ++f)
#pragma unroll
    for (int r = 0; r < 8; ++r) {
      int row = M0 + 16 * wr + r + 8 * h;
      int col = N0 + 64 * wc + 16 * f + ln;
      C[(size_t)row * N + col] = acc[f].f[r];
    }
}

// ---------------------------------------------------------------------------
// Split qkv (fp32) into bf16 q+u, q+v (scale folded), K [b,n,t,hd],
// V transposed [b,n,hd,t].
// ---------------------------------------------------------------------------
__global__ void ep_qkv(const float* __restrict__ qkv,
                       const float* __restrict__ ub, const float* __restrict__ vb,
                       unsigned short* __restrict__ qu,
                       unsigned short* __restrict__ qv,
                       unsigned short* __restrict__ kb,
                       unsigned short* __restrict__ vt) {
  int idx = blockIdx.x * 256 + threadIdx.x;
  if (idx >= BB * NH * TT * HD) return;
  int hd = idx & 63;
  int t = (idx >> 6) & (TT - 1);
  int tmp = idx >> 17;
  int n = tmp % NH, b = tmp / NH;
  size_t row = (size_t)t * BB + b;
  float q = qkv[row * 2304 + n * 64 + hd];
  float kk = qkv[row * 2304 + 768 + n * 64 + hd];
  float vv = qkv[row * 2304 + 1536 + n * 64 + hd];
  size_t o = ((size_t)(b * NH + n) * TT + t) * HD + hd;
  qu[o] = f2bf((q + ub[n * 64 + hd]) * SCALE);
  qv[o] = f2bf((q + vb[n * 64 + hd]) * SCALE);
  kb[o] = f2bf(kk);
  vt[((size_t)(b * NH + n) * HD + hd) * TT + t] = f2bf(vv);
}

// posr[n][d][hd] = bf16(pos[T-1-d][n][hd])  (row-reversed pos projection)
__global__ void ep_pos(const float* __restrict__ pw,
                       unsigned short* __restrict__ posr) {
  int idx = blockIdx.x * 256 + threadIdx.x;
  if (idx >= NH * TT * HD) return;
  int hd = idx & 63;
  int d = (idx >> 6) & (TT - 1);
  int n = idx >> 17;
  posr[((size_t)n * TT + d) * HD + hd] =
      f2bf(pw[(size_t)(TT - 1 - d) * DD + n * 64 + hd]);
}

// ---------------------------------------------------------------------------
// Flash attention with relative position term.
// Grid: (T/128, NH, B); block 256 = 8 waves; wave w owns query rows
// [I0+16w, I0+16w+16). Streams 64-wide key tiles j <= i with double-buffered
// async-DMA staging of K / V^T / pos-band tiles.
//   S[i,j] = Qu_i . K_j  +  Qv_i . posr[i-j]     (scale pre-folded)
// Pos term per wave: WMMA Qv x Band^T (80 cols), sheared via LDS:
//   c = ti - tj + 63.
// ---------------------------------------------------------------------------
__global__ __launch_bounds__(256) void attn_kernel(
    const unsigned short* __restrict__ qu, const unsigned short* __restrict__ qv,
    const unsigned short* __restrict__ kb, const unsigned short* __restrict__ vt,
    const unsigned short* __restrict__ posr, unsigned short* __restrict__ ctx) {
  __shared__ alignas(16) unsigned short Kl[2][64][72];     // K tile [j][hd]
  __shared__ alignas(16) unsigned short Vl[2][64][72];     // V^T tile [hd][t]
  __shared__ alignas(16) unsigned short PosL[2][192][72];  // pos band [c][hd]
  __shared__ alignas(16) unsigned short Pl[8][16][72];     // per-wave P (bf16)
  __shared__ alignas(16) float SposL[8][16][80];           // per-wave pos scores

  const int tid = threadIdx.x, lane = tid & 31, w = tid >> 5;
  const int ln = lane & 15, h = lane >> 4;
  const int b = blockIdx.z, n = blockIdx.y, I0 = blockIdx.x * 128;
  const int iw = I0 + w * 16;
  const size_t headoff = (size_t)(b * NH + n) * TT * HD;   // qu/qv/kb
  const size_t headoffT = (size_t)(b * NH + n) * HD * TT;  // vt
  const size_t posoff = (size_t)n * TT * HD;

  // Issue async DMA for one key tile (K, V^T, pos band) into buffer `buf`.
  auto issue_tile = [&](int jt, int buf) {
    const int J0 = jt * 64;
    const int dstage = I0 - J0 - 63;
#pragma unroll
    for (int it = 0; it < 2; ++it) {
      int idx = tid + 256 * it;
      int r = idx >> 3, c8 = idx & 7;
      async_g2l_b128(lds_addr(&Kl[buf][r][c8 * 8]),
                     kb + headoff + (size_t)(J0 + r) * HD + c8 * 8);
      async_g2l_b128(lds_addr(&Vl[buf][r][c8 * 8]),
                     vt + headoffT + (size_t)r * TT + J0 + c8 * 8);
    }
#pragma unroll
    for (int it = 0; it < 6; ++it) {
      int idx = tid + 256 * it;
      int r = idx >> 3, c8 = idx & 7;
      int d = dstage + r;
      d = d < 0 ? 0 : (d > TT - 1 ? TT - 1 : d);
      async_g2l_b128(lds_addr(&PosL[buf][r][c8 * 8]),
                     posr + posoff + (size_t)d * HD + c8 * 8);
    }
  };

  // Preload Q fragments (A layout: row = ln, K pairs per VGPR).
  Frag16 aQu[2], aQv[2];
#pragma unroll
  for (int ks = 0; ks < 2; ++ks)
#pragma unroll
    for (int v = 0; v < 8; ++v) {
      size_t base = headoff + (size_t)(iw + ln) * HD + kApair(v, h) + 32 * ks;
      aQu[ks].u[v] = *(const unsigned*)(qu + base);
      aQv[ks].u[v] = *(const unsigned*)(qv + base);
    }

  Frag32 O[4];
#pragma unroll
  for (int f = 0; f < 4; ++f)
#pragma unroll
    for (int r = 0; r < 8; ++r) O[f].f[r] = 0.f;
  float mrun[8], lrun[8];
#pragma unroll
  for (int r = 0; r < 8; ++r) { mrun[r] = NEG_BIG; lrun[r] = 0.f; }

  const int nj = I0 / 64 + 2;  // key tiles covering j <= I0+127
  issue_tile(0, 0);            // prologue: prefetch first tile
  for (int jt = 0; jt < nj; ++jt) {
    const int J0 = jt * 64;
    const int buf = jt & 1;
    wait_async0();     // my wave's DMA for `buf` (and older) has landed
    __syncthreads();   // all waves' DMA landed; prior reads of buf^1 retired
    if (jt + 1 < nj) issue_tile(jt + 1, buf ^ 1);  // overlap next tile's DMA

    const bool active = (J0 <= iw + 15);
    Frag32 Sc[4];
    if (active) {
      // ---- content scores: Qu x K^T ----
#pragma unroll
      for (int f = 0; f < 4; ++f) {
#pragma unroll
        for (int r = 0; r < 8; ++r) Sc[f].f[r] = 0.f;
#pragma unroll
        for (int ks = 0; ks < 2; ++ks) {
          Frag16 bf_;
#pragma unroll
          for (int v = 0; v < 8; ++v)
            bf_.u[v] =
                *(const unsigned*)&Kl[buf][16 * f + ln][kBpair(v, h) + 32 * ks];
          Sc[f].v = wmma_bf16(aQu[ks].v, bf_.v, Sc[f].v);
        }
      }
      // ---- pos scores: Qv x Band^T (80 cols), write to wave LDS ----
#pragma unroll
      for (int f = 0; f < 5; ++f) {
        Frag32 sp;
#pragma unroll
        for (int r = 0; r < 8; ++r) sp.f[r] = 0.f;
#pragma unroll
        for (int ks = 0; ks < 2; ++ks) {
          Frag16 bf_;
#pragma unroll
          for (int v = 0; v < 8; ++v)
            bf_.u[v] = *(const unsigned*)&PosL[buf][16 * w + 16 * f + ln]
                                              [kBpair(v, h) + 32 * ks];
          sp.v = wmma_bf16(aQv[ks].v, bf_.v, sp.v);
        }
#pragma unroll
        for (int r = 0; r < 8; ++r) SposL[w][r + 8 * h][ln + 16 * f] = sp.f[r];
      }
    }
    __syncthreads();

    if (active) {
      // ---- combine with shear, causal mask, online softmax ----
      float stot[4][8];
#pragma unroll
      for (int f = 0; f < 4; ++f)
#pragma unroll
        for (int r = 0; r < 8; ++r) {
          int m = r + 8 * h, tj = ln + 16 * f;
          float s = Sc[f].f[r] + SposL[w][m][m - tj + 63];
          if (J0 + tj > iw + m) s = NEG_BIG;  // causal mask
          stot[f][r] = s;
        }
      float corr[8], ps[8];
#pragma unroll
      for (int r = 0; r < 8; ++r) {
        float mx = stot[0][r];
#pragma unroll
        for (int f = 1; f < 4; ++f) mx = fmaxf(mx, stot[f][r]);
        mx = rmax16(mx);
        float nm = fmaxf(mrun[r], mx);
        corr[r] = __expf(mrun[r] - nm);
        mrun[r] = nm;
        ps[r] = 0.f;
      }
#pragma unroll
      for (int f = 0; f < 4; ++f)
#pragma unroll
        for (int r = 0; r < 8; ++r) {
          float p = __expf(stot[f][r] - mrun[r]);
          ps[r] += p;
          Pl[w][r + 8 * h][ln + 16 * f] = f2bf(p);
        }
#pragma unroll
      for (int r = 0; r < 8; ++r) lrun[r] = lrun[r] * corr[r] + rsum16(ps[r]);
#pragma unroll
      for (int f = 0; f < 4; ++f)
#pragma unroll
        for (int r = 0; r < 8; ++r) O[f].f[r] *= corr[r];
    }
    __syncthreads();

    if (active) {
      // ---- O += P x V ----
#pragma unroll
      for (int ks = 0; ks < 2; ++ks) {
        Frag16 af;
#pragma unroll
        for (int v = 0; v < 8; ++v)
          af.u[v] = *(const unsigned*)&Pl[w][ln][kApair(v, h) + 32 * ks];
#pragma unroll
        for (int f = 0; f < 4; ++f) {
          Frag16 bf_;
#pragma unroll
          for (int v = 0; v < 8; ++v)
            bf_.u[v] =
                *(const unsigned*)&Vl[buf][16 * f + ln][kBpair(v, h) + 32 * ks];
          O[f].v = wmma_bf16(af.v, bf_.v, O[f].v);
        }
      }
    }
  }

  // ---- normalize and store context (bf16, [t*B+b, n*64+hd]) ----
#pragma unroll
  for (int f = 0; f < 4; ++f)
#pragma unroll
    for (int r = 0; r < 8; ++r) {
      int m = r + 8 * h, i = iw + m;
      float val = O[f].f[r] / lrun[r];
      ctx[((size_t)i * BB + b) * DD + n * 64 + 16 * f + ln] = f2bf(val);
    }
}

// ---------------------------------------------------------------------------
// Host-side orchestration
// ---------------------------------------------------------------------------
extern "C" void kernel_launch(void* const* d_in, const int* in_sizes, int n_in,
                              void* d_out, int out_size, void* d_ws,
                              size_t ws_size, hipStream_t stream) {
  const float* x = (const float*)d_in[0];        // [T,B,D]
  const float* pos_enc = (const float*)d_in[1];  // [T,D]
  const float* u_bias = (const float*)d_in[2];   // [N,HD]
  const float* v_bias = (const float*)d_in[3];   // [N,HD]
  const float* W_qkv = (const float*)d_in[4];    // [D,3D]
  const float* W_pos = (const float*)d_in[5];    // [D,D]
  const float* W_out = (const float*)d_in[6];    // [D,D]
  // d_in[7] = attn_mask: causal, handled analytically.

  auto au = [](size_t x) { return (x + 255) & ~(size_t)255; };
  char* p = (char*)d_ws;
  auto take = [&](size_t bytes) { char* r = p; p += au(bytes); return r; };

  unsigned short* xb    = (unsigned short*)take((size_t)TT * BB * DD * 2);
  unsigned short* wqkvb = (unsigned short*)take((size_t)DD * 3 * DD * 2);
  unsigned short* wposb = (unsigned short*)take((size_t)DD * DD * 2);
  unsigned short* woutb = (unsigned short*)take((size_t)DD * DD * 2);
  unsigned short* peb   = (unsigned short*)take((size_t)TT * DD * 2);
  unsigned short* qub   = (unsigned short*)take((size_t)BB * NH * TT * HD * 2);
  unsigned short* qvb   = (unsigned short*)take((size_t)BB * NH * TT * HD * 2);
  unsigned short* kbuf  = (unsigned short*)take((size_t)BB * NH * TT * HD * 2);
  unsigned short* vtb   = (unsigned short*)take((size_t)BB * NH * TT * HD * 2);
  unsigned short* posrb = (unsigned short*)take((size_t)NH * TT * HD * 2);
  unsigned short* ctxb  = (unsigned short*)take((size_t)TT * BB * DD * 2);
  float* qkvf = (float*)take((size_t)TT * BB * 3 * DD * 4);
  float* poswf = (float*)take((size_t)TT * DD * 4);
  (void)ws_size; (void)n_in; (void)in_sizes; (void)out_size;

  auto cvt = [&](const float* s, unsigned short* d, int n) {
    cvt_f32_bf16<<<(n + 255) / 256, 256, 0, stream>>>(s, d, n);
  };
  cvt(x, xb, TT * BB * DD);
  cvt(W_qkv, wqkvb, DD * 3 * DD);
  cvt(W_pos, wposb, DD * DD);
  cvt(W_out, woutb, DD * DD);
  cvt(pos_enc, peb, TT * DD);

  // qkv = x @ W_qkv : [4096,768] x [768,2304]
  gemm_bf16<<<dim3(3 * DD / 128, TT * BB / 64), 256, 0, stream>>>(
      xb, DD, wqkvb, 3 * DD, qkvf, TT * BB);
  // pos = pos_enc @ W_pos : [2048,768] x [768,768]
  gemm_bf16<<<dim3(DD / 128, TT / 64), 256, 0, stream>>>(
      peb, DD, wposb, DD, poswf, TT);

  ep_qkv<<<(BB * NH * TT * HD + 255) / 256, 256, 0, stream>>>(
      qkvf, u_bias, v_bias, qub, qvb, kbuf, vtb);
  ep_pos<<<(NH * TT * HD + 255) / 256, 256, 0, stream>>>(poswf, posrb);

  attn_kernel<<<dim3(TT / 128, NH, BB), 256, 0, stream>>>(
      qub, qvb, kbuf, vtb, posrb, ctxb);

  // out = ctx @ W_out : [4096,768] x [768,768] -> fp32 d_out
  gemm_bf16<<<dim3(DD / 128, TT * BB / 64), 256, 0, stream>>>(
      ctxb, DD, woutb, DD, (float*)d_out, TT * BB);
}
